// SwinBlock_89618787598682
// MI455X (gfx1250) — matrix operand order repaired and checked by reading, not compiled, not run on hardware
//
#include <hip/hip_runtime.h>
#include <hip/hip_bf16.h>
#include <math.h>

// ---------------------------------------------------------------------------
// Swin block (shifted window attention + MLP) for MI455X / gfx1250, wave32.
// f16 WMMA (v_wmma_f32_16x16x32_f16) with f32 accumulate for every GEMM.
// ---------------------------------------------------------------------------

typedef _Float16 h16;
typedef __attribute__((ext_vector_type(16))) _Float16 v16h;
typedef __attribute__((ext_vector_type(8)))  float    v8f;

#define CDIM   96
#define HD     32
#define NHEAD  3
#define WSZ    8
#define LTOK   64           // tokens per window (8x8)
#define HIMG   224
#define NWIN_X 28           // 224/8
#define NWIN_PER_B 784      // 28*28
#define NWIN   3136         // 4*784
#define SHIFTV 4            // WS/2
#define MLPH   384          // 4*C

// ------------------------- WMMA tile helpers (wave32) ----------------------
// A tile: 16x32 f16, row-major source with leading dim `ld` (elements).
// Lane l: M = l&15 ; half = l>>4.  VGPR v packs K = 16*half + 2v, 16*half + 2v + 1.
__device__ __forceinline__ v16h load_A16(const h16* base, int ld, int lane) {
  const int m  = lane & 15;
  const int h8 = (lane >> 4) << 4;
  const h16* row = base + m * ld;
  v16h a;
#pragma unroll
  for (int v = 0; v < 8; ++v) {
    const int k = v * 2 + h8;
    a[2 * v]     = row[k];
    a[2 * v + 1] = row[k + 1];
  }
  return a;
}

// B tile: 32x16 f16 where B[k][n] = src[n*ld + k] (src is row-major [N,K],
// i.e. weight matrices in [out,in] layout -> contiguous K per column).
// Lane l: N = l&15 ; kbase = 16*(l>>4). VGPR v packs K = kbase+2v, kbase+2v+1.
__device__ __forceinline__ v16h load_Bt16(const h16* base, int ld, int lane) {
  const int n  = lane & 15;
  const int kb = (lane >> 4) << 4;
  const h16* col = base + n * ld + kb;
  v16h b;
#pragma unroll
  for (int v = 0; v < 8; ++v) {
    b[2 * v]     = col[2 * v];
    b[2 * v + 1] = col[2 * v + 1];
  }
  return b;
}

__device__ __forceinline__ v8f wmma_f16(v16h a, v16h b, v8f c) {
  // D = A*B + C, 16x16x32, f32 accumulate
  return __builtin_amdgcn_wmma_f32_16x16x32_f16(
      /*neg_a=*/false, a, /*neg_b=*/false, b,
      /*c_mod=*/(short)0, c, /*reuse_a=*/false, /*reuse_b=*/false);
}

// ------------------------- weight fp32 -> f16 ------------------------------
// ws layout (halves): [0)=in_proj 27648 | [27648)=out_proj 9216 |
//                     [36864)=mlp_w1 36864 | [73728)=mlp_w2 36864
#define OFF_WIN  0
#define OFF_WOUT 27648
#define OFF_W1   36864
#define OFF_W2   73728
#define NCONV    110592

__global__ void convert_weights(const float* __restrict__ win,
                                const float* __restrict__ wout,
                                const float* __restrict__ w1,
                                const float* __restrict__ w2,
                                h16* __restrict__ dst) {
  int i = blockIdx.x * blockDim.x + threadIdx.x;
  if (i < OFF_WOUT)      dst[i] = (h16)win[i];
  else if (i < OFF_W1)   dst[i] = (h16)wout[i - OFF_WOUT];
  else if (i < OFF_W2)   dst[i] = (h16)w1[i - OFF_W1];
  else if (i < NCONV)    dst[i] = (h16)w2[i - OFF_W2];
}

// ------------------------- fused window attention --------------------------
// One workgroup (256 thr = 8 waves) per window. Does:
//   shifted gather + LN1 -> h (f16 LDS)
//   QKV = h @ Win^T + b (WMMA), V stored transposed
//   per head: scores (WMMA, scaled) -> softmax -> attn@V (WMMA)
//   out = o @ Wout^T + b, scattered back with reverse shift (fp32 -> xnew)
__global__ __launch_bounds__(256) void attn_kernel(
    const float* __restrict__ x,
    const float* __restrict__ ln1w, const float* __restrict__ ln1b,
    const h16*   __restrict__ Win,  const float* __restrict__ bin,
    const h16*   __restrict__ Wout, const float* __restrict__ bout,
    float* __restrict__ xnew) {

  __shared__ h16 sm_h[LTOK * CDIM];    // LN output, later reused as attn output o
  __shared__ h16 sm_q[LTOK * CDIM];
  __shared__ h16 sm_k[LTOK * CDIM];
  __shared__ h16 sm_vt[CDIM * LTOK];   // V transposed: [d][token]
  __shared__ h16 sm_s[LTOK * LTOK];    // scores / attn, one head at a time

  const int tid  = threadIdx.x;
  const int lane = tid & 31;
  // wave index is uniform across the wave: pin it to an SGPR so all tile
  // index math / store-target selection becomes SALU + uniform branches.
  const int wave = __builtin_amdgcn_readfirstlane(tid >> 5);

  const int b   = blockIdx.x / NWIN_PER_B;
  const int rem = blockIdx.x % NWIN_PER_B;
  const int wh  = rem / NWIN_X;
  const int ww  = rem % NWIN_X;

  // ---- gather (roll -4) + LayerNorm1: 4 threads per token, 24 ch each ----
  {
    const int tok  = tid >> 2;
    const int part = tid & 3;
    const int rr   = tok >> 3, cc = tok & 7;
    const int gh = (wh * WSZ + rr + SHIFTV) % HIMG;
    const int gw = (ww * WSZ + cc + SHIFTV) % HIMG;
    const float* src = x + (((size_t)b * HIMG + gh) * HIMG + gw) * CDIM + part * 24;
    float v[24];
    float s = 0.f;
#pragma unroll
    for (int i = 0; i < 24; ++i) { v[i] = src[i]; s += v[i]; }
    s += __shfl_xor(s, 1); s += __shfl_xor(s, 2);
    const float mean = s * (1.0f / CDIM);
    float vs = 0.f;
#pragma unroll
    for (int i = 0; i < 24; ++i) { const float d = v[i] - mean; vs += d * d; }
    vs += __shfl_xor(vs, 1); vs += __shfl_xor(vs, 2);
    const float inv = rsqrtf(vs * (1.0f / CDIM) + 1e-5f);
    h16* hrow = sm_h + tok * CDIM + part * 24;
    const float* w  = ln1w + part * 24;
    const float* bb = ln1b + part * 24;
#pragma unroll
    for (int i = 0; i < 24; ++i)
      hrow[i] = (h16)((v[i] - mean) * inv * w[i] + bb[i]);
  }
  __syncthreads();

  // ---- QKV GEMM: [64,96] x [96,288] -> Q,K (row-major), V (transposed) ----
  // t = wave + 8j  =>  mt = wave&3 is wave-invariant (A tiles stay resident).
  for (int t = wave; t < 72; t += 8) {          // 4 M-tiles x 18 N-tiles
    const int mt = t & 3, nt = t >> 2;
    const int m0 = mt * 16, n0 = nt * 16;
    if (t + 8 < 72)                              // keep weight stream ahead
      __builtin_prefetch(Win + (n0 + 32) * CDIM, 0, 1);
    v8f acc = {};
#pragma unroll
    for (int kt = 0; kt < 3; ++kt) {
      v16h a  = load_A16(sm_h + m0 * CDIM + kt * 32, CDIM, lane);
      v16h bm = load_Bt16(Win + n0 * CDIM + kt * 32, CDIM, lane);
      acc = wmma_f16(a, bm, acc);
    }
    const int n    = n0 + (lane & 15);
    const float bi = bin[n];
    const int mb   = (lane >> 4) * 8;
    if (n0 < CDIM) {                             // uniform (scalar) branch
#pragma unroll
      for (int r = 0; r < 8; ++r)
        sm_q[(m0 + mb + r) * CDIM + n] = (h16)(acc[r] + bi);
    } else if (n0 < 2 * CDIM) {
#pragma unroll
      for (int r = 0; r < 8; ++r)
        sm_k[(m0 + mb + r) * CDIM + (n - CDIM)] = (h16)(acc[r] + bi);
    } else {
#pragma unroll
      for (int r = 0; r < 8; ++r)
        sm_vt[(n - 2 * CDIM) * LTOK + (m0 + mb + r)] = (h16)(acc[r] + bi);
    }
  }
  __syncthreads();

  // ---- per-head attention ----
  for (int head = 0; head < NHEAD; ++head) {
    // scores = (Q Kᵀ) * 1/sqrt(hd)  -> sm_s (f16)
    for (int t = wave; t < 16; t += 8) {        // 4x4 C tiles, K=32 (one WMMA)
      const int mt = t & 3, nt = t >> 2;
      v16h a  = load_A16(sm_q + mt * 16 * CDIM + head * HD, CDIM, lane);
      v16h bm = load_Bt16(sm_k + nt * 16 * CDIM + head * HD, CDIM, lane);
      v8f c = {};
      c = wmma_f16(a, bm, c);
      const int n  = nt * 16 + (lane & 15);
      const int mb = (lane >> 4) * 8;
#pragma unroll
      for (int r = 0; r < 8; ++r)
        sm_s[(mt * 16 + mb + r) * LTOK + n] = (h16)(c[r] * 0.17677669529663687f);
    }
    __syncthreads();

    // softmax over 64 rows, 4 threads/row x 16 cols
    {
      const int row = tid >> 2, part = tid & 3;
      h16* srow = sm_s + row * LTOK + part * 16;
      float vv[16];
      float mx = -1e30f;
#pragma unroll
      for (int i = 0; i < 16; ++i) { vv[i] = (float)srow[i]; mx = fmaxf(mx, vv[i]); }
      mx = fmaxf(mx, __shfl_xor(mx, 1)); mx = fmaxf(mx, __shfl_xor(mx, 2));
      float sum = 0.f;
#pragma unroll
      for (int i = 0; i < 16; ++i) { vv[i] = __expf(vv[i] - mx); sum += vv[i]; }
      sum += __shfl_xor(sum, 1); sum += __shfl_xor(sum, 2);
      const float invs = 1.0f / sum;
#pragma unroll
      for (int i = 0; i < 16; ++i) srow[i] = (h16)(vv[i] * invs);
    }
    __syncthreads();

    // o[:, head*32 : head*32+32] = attn @ V   (A = attn f16, B = Vᵀ rows)
    {
      const int t  = wave;                       // exactly one tile per wave
      const int mt = t & 3, nt = t >> 2;
      v8f c = {};
#pragma unroll
      for (int kt = 0; kt < 2; ++kt) {
        v16h a  = load_A16(sm_s + mt * 16 * LTOK + kt * 32, LTOK, lane);
        v16h bm = load_Bt16(sm_vt + (head * HD + nt * 16) * LTOK + kt * 32, LTOK, lane);
        c = wmma_f16(a, bm, c);
      }
      const int n  = head * HD + nt * 16 + (lane & 15);
      const int mb = (lane >> 4) * 8;
#pragma unroll
      for (int r = 0; r < 8; ++r)
        sm_h[(mt * 16 + mb + r) * CDIM + n] = (h16)c[r];   // sm_h reused as o
    }
    __syncthreads();
  }

  // ---- out projection + reverse shift scatter (fp32) ----
  for (int t = wave; t < 24; t += 8) {          // 4 M-tiles x 6 N-tiles
    const int mt = t % 4, nt = t / 4;
    v8f c = {};
#pragma unroll
    for (int kt = 0; kt < 3; ++kt) {
      v16h a  = load_A16(sm_h + mt * 16 * CDIM + kt * 32, CDIM, lane);
      v16h bm = load_Bt16(Wout + nt * 16 * CDIM + kt * 32, CDIM, lane);
      c = wmma_f16(a, bm, c);
    }
    const int n    = nt * 16 + (lane & 15);
    const float bi = bout[n];
    const int mb   = (lane >> 4) * 8;
#pragma unroll
    for (int r = 0; r < 8; ++r) {
      const int m  = mt * 16 + mb + r;
      const int rr = m >> 3, cc = m & 7;
      const int gh = (wh * WSZ + rr + SHIFTV) % HIMG;
      const int gw = (ww * WSZ + cc + SHIFTV) % HIMG;
      xnew[(((size_t)b * HIMG + gh) * HIMG + gw) * CDIM + n] = c[r] + bi;
    }
  }
}

// ------------------------- fused LN2 + MLP + residual ----------------------
// 64 tokens per block; fully in-place on xio (each block owns its rows).
__global__ __launch_bounds__(256) void mlp_kernel(
    float* __restrict__ xio,
    const float* __restrict__ ln2w, const float* __restrict__ ln2b,
    const h16*   __restrict__ W1,   const float* __restrict__ b1,
    const h16*   __restrict__ W2,   const float* __restrict__ b2) {

  __shared__ h16 sm_h2[LTOK * CDIM];     // LN2 output (f16)
  __shared__ h16 sm_act[LTOK * MLPH];    // GELU(h2 @ W1ᵀ + b1) (f16)

  const int tid  = threadIdx.x;
  const int lane = tid & 31;
  const int wave = __builtin_amdgcn_readfirstlane(tid >> 5);
  const int T0   = blockIdx.x * LTOK;

  // ---- LayerNorm2: 4 threads/token ----
  {
    const int tok  = tid >> 2;
    const int part = tid & 3;
    const float* src = xio + ((size_t)(T0 + tok)) * CDIM + part * 24;
    float v[24];
    float s = 0.f;
#pragma unroll
    for (int i = 0; i < 24; ++i) { v[i] = src[i]; s += v[i]; }
    s += __shfl_xor(s, 1); s += __shfl_xor(s, 2);
    const float mean = s * (1.0f / CDIM);
    float vs = 0.f;
#pragma unroll
    for (int i = 0; i < 24; ++i) { const float d = v[i] - mean; vs += d * d; }
    vs += __shfl_xor(vs, 1); vs += __shfl_xor(vs, 2);
    const float inv = rsqrtf(vs * (1.0f / CDIM) + 1e-5f);
    h16* hrow = sm_h2 + tok * CDIM + part * 24;
    const float* w  = ln2w + part * 24;
    const float* bb = ln2b + part * 24;
#pragma unroll
    for (int i = 0; i < 24; ++i)
      hrow[i] = (h16)((v[i] - mean) * inv * w[i] + bb[i]);
  }
  __syncthreads();

  // ---- GEMM1: [64,96] x [96,384] + exact GELU -> sm_act (f16) ----
  for (int t = wave; t < 96; t += 8) {          // 4 M-tiles x 24 N-tiles
    const int mt = t % 4, nt = t / 4;
    if (t + 8 < 96)
      __builtin_prefetch(W1 + (nt * 16 + 32) * CDIM, 0, 1);
    v8f c = {};
#pragma unroll
    for (int kt = 0; kt < 3; ++kt) {
      v16h a  = load_A16(sm_h2 + mt * 16 * CDIM + kt * 32, CDIM, lane);
      v16h bm = load_Bt16(W1 + nt * 16 * CDIM + kt * 32, CDIM, lane);
      c = wmma_f16(a, bm, c);
    }
    const int n    = nt * 16 + (lane & 15);
    const float bi = b1[n];
    const int mb   = (lane >> 4) * 8;
#pragma unroll
    for (int r = 0; r < 8; ++r) {
      const float xv = c[r] + bi;
      const float g  = 0.5f * xv * (1.0f + erff(xv * 0.70710678118654752f));
      sm_act[(mt * 16 + mb + r) * MLPH + n] = (h16)g;
    }
  }
  __syncthreads();

  // ---- GEMM2: [64,384] x [384,96] + bias + residual, in place ----
  for (int t = wave; t < 24; t += 8) {          // 4 M-tiles x 6 N-tiles
    const int mt = t % 4, nt = t / 4;
    v8f c = {};
#pragma unroll
    for (int kt = 0; kt < 12; ++kt) {
      v16h a  = load_A16(sm_act + mt * 16 * MLPH + kt * 32, MLPH, lane);
      v16h bm = load_Bt16(W2 + nt * 16 * MLPH + kt * 32, MLPH, lane);
      c = wmma_f16(a, bm, c);
    }
    const int n    = nt * 16 + (lane & 15);
    const float bi = b2[n];
    const int mb   = (lane >> 4) * 8;
#pragma unroll
    for (int r = 0; r < 8; ++r) {
      const int m = mt * 16 + mb + r;
      const size_t addr = ((size_t)(T0 + m)) * CDIM + n;
      const float res = xio[addr];              // residual (read-then-write, same lane)
      xio[addr] = res + c[r] + bi;
    }
  }
}

// ---------------------------------------------------------------------------
extern "C" void kernel_launch(void* const* d_in, const int* in_sizes, int n_in,
                              void* d_out, int out_size, void* d_ws, size_t ws_size,
                              hipStream_t stream) {
  const float* x      = (const float*)d_in[0];
  const float* ln1w   = (const float*)d_in[1];
  const float* ln1b   = (const float*)d_in[2];
  const float* inw    = (const float*)d_in[3];
  const float* inb    = (const float*)d_in[4];
  const float* outw   = (const float*)d_in[5];
  const float* outb   = (const float*)d_in[6];
  const float* ln2w   = (const float*)d_in[7];
  const float* ln2b   = (const float*)d_in[8];
  const float* w1     = (const float*)d_in[9];
  const float* b1     = (const float*)d_in[10];
  const float* w2     = (const float*)d_in[11];
  const float* b2     = (const float*)d_in[12];
  float* out = (float*)d_out;
  h16*   wh  = (h16*)d_ws;

  convert_weights<<<(NCONV + 255) / 256, 256, 0, stream>>>(inw, outw, w1, w2, wh);

  attn_kernel<<<NWIN, 256, 0, stream>>>(
      x, ln1w, ln1b,
      wh + OFF_WIN, inb, wh + OFF_WOUT, outb,
      out);

  mlp_kernel<<<NWIN, 256, 0, stream>>>(     // 200704 tokens / 64 = 3136 blocks
      out, ln2w, ln2b,
      wh + OFF_W1, b1, wh + OFF_W2, b2);
}
